// SymbolicGraphEncoder_38543036514920
// MI455X (gfx1250) — matrix operand order repaired and checked by reading, not compiled, not run on hardware
//
#include <hip/hip_runtime.h>

// ---------------------------------------------------------------------------
// 2-layer GCN + mean pool for MI455X (gfx1250, wave32, WMMA).
// Dense transforms: v_wmma_f32_16x16x32_f16, dual accumulators per wave to
// hide the WMMA->WMMA XDL hazard, self-loop+bias fused into the epilogue.
// Sparse aggregation: wave-uniform edge indices (scalar loads) + float4
// gathers + global_atomic_add_f32 scatters.
// ---------------------------------------------------------------------------

typedef __attribute__((ext_vector_type(16))) _Float16 v16h;
typedef __attribute__((ext_vector_type(8)))  float    v8f;

constexpr int Nn = 40000;   // nodes
constexpr int Ee = 640000;  // edges
constexpr int Hh = 128;     // feature width (IN == H == 128)
constexpr int Gg = 64;      // graphs

union Frag16 { v16h v; uint4 q[2]; };

// ---- degree / normalization ------------------------------------------------
__global__ void k_fill_deg(float* __restrict__ deg) {
  int i = blockIdx.x * 256 + threadIdx.x;
  if (i < Nn) deg[i] = 1.0f;                 // self-loop counts as 1
}

__global__ void k_count(const int* __restrict__ dst, float* __restrict__ deg) {
  int e = blockIdx.x * 256 + threadIdx.x;
  if (e < Ee) atomicAdd(&deg[dst[e]], 1.0f);
}

__global__ void k_rsqrt(float* __restrict__ deg) {
  int i = blockIdx.x * 256 + threadIdx.x;
  if (i < Nn) deg[i] = rsqrtf(deg[i]);       // deg buffer becomes dinv
}

// ---- pack W[128][128] f32 -> per-lane WMMA B fragments (f16) --------------
// Layout: Wp[tile(8)][kstep(4)][lane(32)][e(16)] with
//   column n = tile*16 + (lane&15),  K = kstep*32 + 16*(lane>>4) + e
__global__ void k_prepack(const float* __restrict__ W, _Float16* __restrict__ Wp) {
  int idx = blockIdx.x * 256 + threadIdx.x;  // 16384 total
  int e    = idx & 15;
  int lane = (idx >> 4) & 31;
  int ks   = (idx >> 9) & 3;
  int tile = idx >> 11;
  int n = tile * 16 + (lane & 15);
  int K = ks * 32 + ((lane >> 4) << 4) + e;
  Wp[idx] = (_Float16)W[K * Hh + n];
}

// ---- WMMA GEMM + fused self-loop/bias epilogue ----------------------------
// Block: 256 threads (8 waves), covers 32 rows x 128 cols. Wave w owns cols
// [16w,16w+16) with TWO 16x16 accumulators (rows m0..m0+15 and m0+16..m0+31)
// sharing each B fragment -> interleaved WMMAs hide the D->C RAW hazard.
//   Y   = (relu?)X @ W              (raw transform, needed by edge scatter)
//   AGG = Y * dinv^2 + bias         (self-loop + bias init for scatter-add)
template <bool RELU>
__global__ void k_gemm_wmma(const float* __restrict__ X,
                            const _Float16* __restrict__ Wp,
                            const float* __restrict__ dinv,
                            const float* __restrict__ bias,
                            float* __restrict__ Y,
                            float* __restrict__ AGG) {
  __shared__ alignas(16) _Float16 ldsA[32 * 128];
  const int tid = threadIdx.x;
  const int m0 = blockIdx.x * 32;            // N = 40000 = 1250 * 32, no tail

  // Stage 32x128 A-slab into LDS as f16 (fused ReLU for layer 2).
  for (int i = tid; i < 32 * 128; i += 256) {
    float v = X[(size_t)(m0 + (i >> 7)) * Hh + (i & 127)];
    if (RELU) v = fmaxf(v, 0.0f);
    ldsA[i] = (_Float16)v;
  }
  __syncthreads();

  const int wv = tid >> 5;
  const int lane = tid & 31;
  const int row = lane & 15;                 // A-matrix M index (tile-local)
  const int klane = (lane >> 4) << 4;        // 0 (lanes 0-15) / 16 (lanes 16-31)

  v8f acc0 = {}, acc1 = {};
  const uint4* bp = (const uint4*)(Wp + ((size_t)(wv * 4) * 32 + lane) * 16);

#pragma unroll
  for (int ks = 0; ks < 4; ++ks) {
    Frag16 a0, a1, b;
    const int kbase = ks * 32 + klane;
    // k order used here (same as Wp): lane-half covers contiguous K run [kbase,kbase+16)
    a0.q[0] = *(const uint4*)&ldsA[row * 128 + kbase];
    a0.q[1] = *(const uint4*)&ldsA[row * 128 + kbase + 8];
    a1.q[0] = *(const uint4*)&ldsA[(row + 16) * 128 + kbase];
    a1.q[1] = *(const uint4*)&ldsA[(row + 16) * 128 + kbase + 8];
    b.q[0] = bp[0];
    b.q[1] = bp[1];
    bp += 64;                                // 512 halves = next kstep
    acc0 = __builtin_amdgcn_wmma_f32_16x16x32_f16(
        false, a0.v, false, b.v, (short)0, acc0, false, false);
    acc1 = __builtin_amdgcn_wmma_f32_16x16x32_f16(
        false, a1.v, false, b.v, (short)0, acc1, false, false);
  }

  // C/D layout: lane -> N = lane&15, acc[r] -> M = r + 8*(lane>>4)
  const int n = wv * 16 + (lane & 15);
  const float bn = bias[n];
  const int mb = m0 + ((lane >> 4) << 3);
#pragma unroll
  for (int r = 0; r < 8; ++r) {
    const int m = mb + r;
    const float di = dinv[m];
    const float v = acc0[r];
    Y[(size_t)m * Hh + n] = v;
    AGG[(size_t)m * Hh + n] = v * di * di + bn;
    const int m2 = m + 16;
    const float di2 = dinv[m2];
    const float v2 = acc1[r];
    Y[(size_t)m2 * Hh + n] = v2;
    AGG[(size_t)m2 * Hh + n] = v2 * di2 * di2 + bn;
  }
}

// ---- edge scatter: agg[dst] += xw[src] * dinv[src]*dinv[dst] --------------
// One wave per edge (wave-uniform e -> scalar s_loads for src/dst/dinv),
// 4 features per lane: float4 gather + 4 f32 atomics.
__global__ void k_scatter(const float* __restrict__ xw,
                          const int* __restrict__ src,
                          const int* __restrict__ dst,
                          const float* __restrict__ dinv,
                          float* __restrict__ agg) {
  const int e = blockIdx.x * 8 + (threadIdx.x >> 5);  // 80000*8 == E exactly
  const int f = (threadIdx.x & 31) * 4;
  const int s = src[e];
  const int d = dst[e];
  const float c = dinv[s] * dinv[d];
  const float4 v = *(const float4*)&xw[(size_t)s * Hh + f];
  float* ap = &agg[(size_t)d * Hh + f];
  atomicAdd(ap + 0, v.x * c);
  atomicAdd(ap + 1, v.y * c);
  atomicAdd(ap + 2, v.z * c);
  atomicAdd(ap + 3, v.w * c);
}

// ---- pooling --------------------------------------------------------------
__global__ void k_zero(float* __restrict__ out, float* __restrict__ cnt) {
  int i = blockIdx.x * 256 + threadIdx.x;
  if (i < Gg * Hh) out[i] = 0.0f;
  if (i < Gg) cnt[i] = 0.0f;
}

__global__ void k_cnt(const int* __restrict__ batch, float* __restrict__ cnt) {
  int i = blockIdx.x * 256 + threadIdx.x;
  if (i < Nn) atomicAdd(&cnt[batch[i]], 1.0f);
}

__global__ void k_poolsum(const float* __restrict__ h,
                          const int* __restrict__ batch,
                          float* __restrict__ out) {
  const int node = blockIdx.x * 8 + (threadIdx.x >> 5);  // 5000*8 == N exactly
  const int f = (threadIdx.x & 31) * 4;
  const int g = batch[node];                              // wave-uniform
  const float4 v = *(const float4*)&h[(size_t)node * Hh + f];
  float* op = &out[g * Hh + f];
  atomicAdd(op + 0, fmaxf(v.x, 0.0f));   // final ReLU fused here
  atomicAdd(op + 1, fmaxf(v.y, 0.0f));
  atomicAdd(op + 2, fmaxf(v.z, 0.0f));
  atomicAdd(op + 3, fmaxf(v.w, 0.0f));
}

__global__ void k_div(float* __restrict__ out, const float* __restrict__ cnt) {
  int i = blockIdx.x * 256 + threadIdx.x;
  if (i < Gg * Hh) out[i] /= fmaxf(cnt[i >> 7], 1.0f);
}

// ---------------------------------------------------------------------------
extern "C" void kernel_launch(void* const* d_in, const int* in_sizes, int n_in,
                              void* d_out, int out_size, void* d_ws, size_t ws_size,
                              hipStream_t stream) {
  (void)in_sizes; (void)n_in; (void)out_size; (void)ws_size;
  const float* x     = (const float*)d_in[0];              // [N,128]
  const int*   eidx  = (const int*)d_in[1];                // [2,E]
  const int*   batch = (const int*)d_in[2];                // [N]
  // d_in[3] = num_graphs (compile-time 64)
  const float* W1 = (const float*)d_in[4];
  const float* b1 = (const float*)d_in[5];
  const float* W2 = (const float*)d_in[6];
  const float* b2 = (const float*)d_in[7];
  const int* src = eidx;
  const int* dst = eidx + Ee;
  float* out = (float*)d_out;                              // [G,128]

  // workspace layout (all offsets 16B-aligned)
  float* xw   = (float*)d_ws;            // N*H
  float* agg  = xw + (size_t)Nn * Hh;    // N*H
  float* dinv = agg + (size_t)Nn * Hh;   // N
  float* cnt  = dinv + Nn;               // G
  _Float16* W1p = (_Float16*)(cnt + Gg); // H*H halves
  _Float16* W2p = W1p + Hh * Hh;         // H*H halves

  const int B = 256;
  const int nbN    = (Nn + B - 1) / B;         // 157
  const int nbE    = (Ee + B - 1) / B;         // 2500
  const int nbE8   = Ee / 8;                   // 80000 (1 wave per edge)
  const int nbN8   = Nn / 8;                   // 5000  (1 wave per node)
  const int nbW    = (Hh * Hh) / B;            // 64 (prepack: 16384 elems)
  const int nbG    = (Gg * Hh + B - 1) / B;    // 32
  const int nbGemm = Nn / 32;                  // 1250

  // normalization
  k_fill_deg<<<nbN, B, 0, stream>>>(dinv);
  k_count<<<nbE, B, 0, stream>>>(dst, dinv);
  k_rsqrt<<<nbN, B, 0, stream>>>(dinv);

  // pack weights into WMMA B-fragment layout
  k_prepack<<<nbW, B, 0, stream>>>(W1, W1p);
  k_prepack<<<nbW, B, 0, stream>>>(W2, W2p);

  // layer 1: xw = x @ W1 ; agg = xw*dinv^2 + b1 (fused) ; scatter edges
  k_gemm_wmma<false><<<nbGemm, B, 0, stream>>>(x, W1p, dinv, b1, xw, agg);
  k_scatter<<<nbE8, B, 0, stream>>>(xw, src, dst, dinv, agg);

  // layer 2: xw = relu(agg) @ W2 ; agg = xw*dinv^2 + b2 (fused, in-place
  // safe: each block stages its exclusive 32 rows before storing) ; scatter
  k_gemm_wmma<true><<<nbGemm, B, 0, stream>>>(agg, W2p, dinv, b2, xw, agg);
  k_scatter<<<nbE8, B, 0, stream>>>(xw, src, dst, dinv, agg);

  // global mean pool with fused final ReLU
  k_zero<<<nbG, B, 0, stream>>>(out, cnt);
  k_cnt<<<nbN, B, 0, stream>>>(batch, cnt);
  k_poolsum<<<nbN8, B, 0, stream>>>(agg, batch, out);
  k_div<<<nbG, B, 0, stream>>>(out, cnt);
}
